// MultiHeadAttention_86466281603669
// MI455X (gfx1250) — matrix operand order repaired and checked by reading, not compiled, not run on hardware
//
#include <hip/hip_runtime.h>

// ---------------------------------------------------------------------------
// MHA forward for MI455X (gfx1250, wave32, WMMA bf16 16x16x32, f32 accum)
// B=2 S=2048 D=1024 H=16 HD=64
// ---------------------------------------------------------------------------

#define B_  2
#define S_  2048
#define D_  1024
#define H_  16
#define HD_ 64
#define SCALE_ 0.125f   // 1/sqrt(64)

typedef __attribute__((ext_vector_type(16))) __bf16 v16bf;
typedef __attribute__((ext_vector_type(8)))  __bf16 v8bf;
typedef __attribute__((ext_vector_type(8)))  float  v8f;
typedef __attribute__((ext_vector_type(4)))  float  f32x4;
typedef int i32x4v __attribute__((vector_size(16)));   // matches builtin param type

// ---- optional gfx1250 async global->LDS path (ASYNCcnt) -------------------
#if defined(__has_builtin)
#if __has_builtin(__builtin_amdgcn_global_load_async_to_lds_b128) && \
    __has_builtin(__builtin_amdgcn_s_wait_asynccnt)
#define HAVE_ASYNC_LDS 1
#endif
#endif
#ifndef HAVE_ASYNC_LDS
#define HAVE_ASYNC_LDS 0
#endif

#if HAVE_ASYNC_LDS
static __device__ __forceinline__ void async_g2l_16B(const void* g, void* l) {
    // generic LDS address: aperture in bits 63:32, LDS byte offset in bits 31:0
    __builtin_amdgcn_global_load_async_to_lds_b128(
        (__attribute__((address_space(1))) i32x4v*)(unsigned long long)g,
        (__attribute__((address_space(3))) i32x4v*)(unsigned)(unsigned long long)l,
        0, 0);
}
#endif

static __device__ __forceinline__ v8f wmma_bf16(v16bf a, v16bf b, v8f c) {
    return __builtin_amdgcn_wmma_f32_16x16x32_bf16(
        false, a, false, b, (short)0, c, false, false);
}

static __device__ __forceinline__ v16bf mk16(v8bf lo, v8bf hi) {
    v16bf r;
#pragma unroll
    for (int i = 0; i < 8; ++i) { r[i] = lo[i]; r[i + 8] = hi[i]; }
    return r;
}

static __device__ __forceinline__ v8bf cvt8(const float* p) {
    f32x4 a = *(const f32x4*)p;
    f32x4 b = *(const f32x4*)(p + 4);
    v8bf r;
    r[0] = (__bf16)a[0]; r[1] = (__bf16)a[1]; r[2] = (__bf16)a[2]; r[3] = (__bf16)a[3];
    r[4] = (__bf16)b[0]; r[5] = (__bf16)b[1]; r[6] = (__bf16)b[2]; r[7] = (__bf16)b[3];
    return r;
}

// ---- A fragment (16x32, MxK): lane<16: row=lane, K 0..15;
//                               lane>=16: row=lane-16, K 16..31
static __device__ __forceinline__ v16bf a_frag_bf16(const __bf16* A, int lda, int mbase, int k0) {
    int lane = threadIdx.x & 31, half = lane >> 4, nl = lane & 15;
    const __bf16* row = A + (size_t)(mbase + nl) * lda + k0 + half * 16;
    return mk16(*(const v8bf*)row, *(const v8bf*)(row + 8));
}

// ---- B fragment (32x16, KxN) from row-major Wt[N][K]  (B = Wt^T):
//      lane<16: col=lane, K 0..15;  lane>=16: col=lane-16, K 16..31
static __device__ __forceinline__ v16bf b_frag_bf16T(const __bf16* Wt, int ldw, int nbase, int k0) {
    int lane = threadIdx.x & 31, half = lane >> 4, nl = lane & 15;
    const __bf16* row = Wt + (size_t)(nbase + nl) * ldw + k0 + half * 16;
    return mk16(*(const v8bf*)row, *(const v8bf*)(row + 8));
}

// ===========================================================================
// Kernel 0: f32 -> bf16 elementwise convert (x and the four weight matrices)
// ===========================================================================
__global__ void __launch_bounds__(256)
cvt_bf16(const float* __restrict__ in, __bf16* __restrict__ out, int n8) {
    int i = blockIdx.x * 256 + threadIdx.x;
    if (i < n8) {
        *(v8bf*)(out + (size_t)i * 8) = cvt8(in + (size_t)i * 8);
    }
}

// ===========================================================================
// Kernel 1: QKV projection.  C[m,n] = sum_k x[m,k] * W[n,k]  (x @ W^T)
// grid (N/128, M/64, 3), block 128 (4 waves, each 32x64 via 2x4 WMMA tiles).
// Output bf16 in [B,H,S,HD].
// ===========================================================================
__global__ void __launch_bounds__(128)
qkv_proj(const __bf16* __restrict__ x,
         const __bf16* __restrict__ Wq, const __bf16* __restrict__ Wk,
         const __bf16* __restrict__ Wv,
         __bf16* __restrict__ q, __bf16* __restrict__ k, __bf16* __restrict__ v)
{
    const int z = blockIdx.z;
    const __bf16* W   = (z == 0) ? Wq : (z == 1) ? Wk : Wv;
    __bf16*       out = (z == 0) ? q  : (z == 1) ? k  : v;

    const int wave = threadIdx.x >> 5, lane = threadIdx.x & 31;
    const int half = lane >> 4, nl = lane & 15;
    const int wm = blockIdx.y * 64  + (wave >> 1) * 32;
    const int wn = blockIdx.x * 128 + (wave & 1) * 64;

    v8f zero = {};
    v8f acc[2][4];
#pragma unroll
    for (int i = 0; i < 2; ++i)
#pragma unroll
        for (int j = 0; j < 4; ++j) acc[i][j] = zero;

    for (int k0 = 0; k0 < D_; k0 += 32) {
        __builtin_prefetch(x + (size_t)wm * D_ + k0 + 128, 0, 1);
        v16bf a0 = a_frag_bf16(x, D_, wm,      k0);
        v16bf a1 = a_frag_bf16(x, D_, wm + 16, k0);
#pragma unroll
        for (int j = 0; j < 4; ++j) {
            v16bf bj = b_frag_bf16T(W, D_, wn + j * 16, k0);
            acc[0][j] = wmma_bf16(a0, bj, acc[0][j]);
            acc[1][j] = wmma_bf16(a1, bj, acc[1][j]);
        }
    }

    // C row m = b*S+s, col n = h*HD+hd  ->  out[b][h][s][hd]
#pragma unroll
    for (int mi = 0; mi < 2; ++mi)
#pragma unroll
        for (int nj = 0; nj < 4; ++nj)
#pragma unroll
            for (int r = 0; r < 8; ++r) {
                int m = wm + mi * 16 + half * 8 + r;
                int n = wn + nj * 16 + nl;
                int bb = m >> 11, ss = m & (S_ - 1);
                int hh = n >> 6,  hd = n & (HD_ - 1);
                out[(((size_t)bb * H_ + hh) * S_ + ss) * HD_ + hd] = (__bf16)acc[mi][nj][r];
            }
}

// ===========================================================================
// Kernel 2: causal flash attention.  grid (S/64, H, B), block 128.
// ===========================================================================
__global__ void __launch_bounds__(128)
attn(const __bf16* __restrict__ Q, const __bf16* __restrict__ K,
     const __bf16* __restrict__ V, __bf16* __restrict__ Ctx)
{
    __shared__ __bf16 sK [32][72];        // K tile  [key][hd]   (+8 pad)
    __shared__ __bf16 sVt[64][40];        // V tile transposed [hd][key] (+8 pad)
    __shared__ __bf16 sP [4][16][40];     // per-wave P tile [row][key] (+8 pad)

    const int b = blockIdx.z, h = blockIdx.y;
    const int qblock = blockIdx.x * 64;
    const int wave = threadIdx.x >> 5, lane = threadIdx.x & 31;
    const int half = lane >> 4, nl = lane & 15;
    const int qbase = qblock + wave * 16;

    const size_t headoff = (size_t)(b * H_ + h) * S_ * HD_;
    const __bf16* Qh = Q + headoff;
    const __bf16* Kh = K + headoff;
    const __bf16* Vh = V + headoff;

    v16bf aq0 = a_frag_bf16(Qh, HD_, qbase, 0);
    v16bf aq1 = a_frag_bf16(Qh, HD_, qbase, 32);

    v8f zero = {};
    v8f co[4]; co[0] = zero; co[1] = zero; co[2] = zero; co[3] = zero;
    float mrow[8], lrow[8];
#pragma unroll
    for (int r = 0; r < 8; ++r) { mrow[r] = -3.0e38f; lrow[r] = 0.0f; }

    const int nsteps = (qblock + 64) / 32;
    for (int st = 0; st < nsteps; ++st) {
        const int kb = st * 32;
        __syncthreads();
        {   // stage K (async DMA to LDS when available) + transposed V tile
            int t = threadIdx.x;
            int kr = t >> 2, cc = (t & 3) * 16;
            const __bf16* ksrc = Kh + (size_t)(kb + kr) * HD_ + cc;
#if HAVE_ASYNC_LDS
            async_g2l_16B(ksrc,     &sK[kr][cc]);
            async_g2l_16B(ksrc + 8, &sK[kr][cc + 8]);
#else
            v8bf k0v = *(const v8bf*)ksrc, k1v = *(const v8bf*)(ksrc + 8);
            *(v8bf*)&sK[kr][cc]     = k0v;
            *(v8bf*)&sK[kr][cc + 8] = k1v;
#endif
            const __bf16* vsrc = Vh + (size_t)(kb + kr) * HD_ + cc;
            v8bf v0 = *(const v8bf*)vsrc, v1 = *(const v8bf*)(vsrc + 8);
#pragma unroll
            for (int i = 0; i < 8; ++i) {
                sVt[cc + i][kr]     = v0[i];
                sVt[cc + 8 + i][kr] = v1[i];
            }
        }
#if HAVE_ASYNC_LDS
        __builtin_amdgcn_s_wait_asynccnt(0);
#endif
        __syncthreads();

        // --- scores: 16 queries x 32 keys (two 16x16 tiles, K-dim = HD = 2x32)
        v8f s0 = zero, s1 = zero;
        {
            v16bf bk;
            bk = mk16(*(const v8bf*)&sK[nl][half * 16],
                      *(const v8bf*)&sK[nl][half * 16 + 8]);
            s0 = wmma_bf16(aq0, bk, s0);
            bk = mk16(*(const v8bf*)&sK[nl][32 + half * 16],
                      *(const v8bf*)&sK[nl][32 + half * 16 + 8]);
            s0 = wmma_bf16(aq1, bk, s0);
            bk = mk16(*(const v8bf*)&sK[16 + nl][half * 16],
                      *(const v8bf*)&sK[16 + nl][half * 16 + 8]);
            s1 = wmma_bf16(aq0, bk, s1);
            bk = mk16(*(const v8bf*)&sK[16 + nl][32 + half * 16],
                      *(const v8bf*)&sK[16 + nl][32 + half * 16 + 8]);
            s1 = wmma_bf16(aq1, bk, s1);
        }

        // --- scale, causal mask, online softmax
#pragma unroll
        for (int r = 0; r < 8; ++r) {
            int qi = qbase + half * 8 + r;
            int k0i = kb + nl, k1i = kb + 16 + nl;
            float x0 = s0[r] * SCALE_; if (k0i > qi) x0 = -3.0e38f;
            float x1 = s1[r] * SCALE_; if (k1i > qi) x1 = -3.0e38f;
            float mx = fmaxf(x0, x1);
            mx = fmaxf(mx, __shfl_xor(mx, 8, 16));
            mx = fmaxf(mx, __shfl_xor(mx, 4, 16));
            mx = fmaxf(mx, __shfl_xor(mx, 2, 16));
            mx = fmaxf(mx, __shfl_xor(mx, 1, 16));
            float mnew = fmaxf(mrow[r], mx);
            float p0 = __expf(x0 - mnew);
            float p1 = __expf(x1 - mnew);
            float alpha = __expf(mrow[r] - mnew);
            mrow[r] = mnew;
            float ps = p0 + p1;
            ps += __shfl_xor(ps, 8, 16);
            ps += __shfl_xor(ps, 4, 16);
            ps += __shfl_xor(ps, 2, 16);
            ps += __shfl_xor(ps, 1, 16);
            lrow[r] = lrow[r] * alpha + ps;
#pragma unroll
            for (int t = 0; t < 4; ++t) co[t][r] *= alpha;
            sP[wave][half * 8 + r][nl]      = (__bf16)p0;
            sP[wave][half * 8 + r][16 + nl] = (__bf16)p1;
        }

        // --- ctx += P * V
        v16bf ap = mk16(*(const v8bf*)&sP[wave][nl][half * 16],
                        *(const v8bf*)&sP[wave][nl][half * 16 + 8]);
#pragma unroll
        for (int t = 0; t < 4; ++t) {
            v16bf bv = mk16(*(const v8bf*)&sVt[t * 16 + nl][half * 16],
                            *(const v8bf*)&sVt[t * 16 + nl][half * 16 + 8]);
            co[t] = wmma_bf16(ap, bv, co[t]);
        }
    }

    // --- normalize, write ctx as [B,S,H,HD]
#pragma unroll
    for (int r = 0; r < 8; ++r) {
        float inv = 1.0f / lrow[r];
        int qi = qbase + half * 8 + r;
        size_t base = (((size_t)b * S_ + qi) * H_ + h) * HD_ + nl;
#pragma unroll
        for (int t = 0; t < 4; ++t)
            Ctx[base + t * 16] = (__bf16)(co[t][r] * inv);
    }
}

// ===========================================================================
// Kernel 3: output projection.  out[m,n] = sum_k ctx[m,k]*Wo[n,k] + bo[n]
// grid (N/128, M/64), block 128, 2x4 WMMA tiles per wave.
// ===========================================================================
__global__ void __launch_bounds__(128)
out_proj(const __bf16* __restrict__ ctx, const __bf16* __restrict__ Wo,
         const float* __restrict__ bo, float* __restrict__ out)
{
    const int wave = threadIdx.x >> 5, lane = threadIdx.x & 31;
    const int half = lane >> 4, nl = lane & 15;
    const int wm = blockIdx.y * 64  + (wave >> 1) * 32;
    const int wn = blockIdx.x * 128 + (wave & 1) * 64;

    v8f zero = {};
    v8f acc[2][4];
#pragma unroll
    for (int i = 0; i < 2; ++i)
#pragma unroll
        for (int j = 0; j < 4; ++j) acc[i][j] = zero;

    for (int k0 = 0; k0 < D_; k0 += 32) {
        __builtin_prefetch(ctx + (size_t)wm * D_ + k0 + 128, 0, 1);
        v16bf a0 = a_frag_bf16(ctx, D_, wm,      k0);
        v16bf a1 = a_frag_bf16(ctx, D_, wm + 16, k0);
#pragma unroll
        for (int j = 0; j < 4; ++j) {
            v16bf bj = b_frag_bf16T(Wo, D_, wn + j * 16, k0);
            acc[0][j] = wmma_bf16(a0, bj, acc[0][j]);
            acc[1][j] = wmma_bf16(a1, bj, acc[1][j]);
        }
    }

#pragma unroll
    for (int mi = 0; mi < 2; ++mi)
#pragma unroll
        for (int nj = 0; nj < 4; ++nj)
#pragma unroll
            for (int r = 0; r < 8; ++r) {
                int m = wm + mi * 16 + half * 8 + r;
                int n = wn + nj * 16 + nl;
                out[(size_t)m * D_ + n] = acc[mi][nj][r] + bo[n];
            }
}

// ===========================================================================
extern "C" void kernel_launch(void* const* d_in, const int* in_sizes, int n_in,
                              void* d_out, int out_size, void* d_ws, size_t ws_size,
                              hipStream_t stream) {
    const float* x  = (const float*)d_in[0];
    const float* Wq = (const float*)d_in[1];
    const float* Wk = (const float*)d_in[2];
    const float* Wv = (const float*)d_in[3];
    const float* Wo = (const float*)d_in[4];
    const float* bo = (const float*)d_in[5];
    float* out = (float*)d_out;

    const size_t TOK = (size_t)B_ * S_ * D_;   // 4,194,304
    const size_t WSZ = (size_t)D_ * D_;        // 1,048,576
    __bf16* qb  = (__bf16*)d_ws;               // Q  [B,H,S,HD]
    __bf16* kb  = qb + TOK;                    // K  [B,H,S,HD]
    __bf16* vb  = kb + TOK;                    // V  [B,H,S,HD]
    __bf16* cb  = vb + TOK;                    // ctx[B,S,H,HD]
    __bf16* xb  = cb + TOK;                    // x  bf16
    __bf16* wqb = xb + TOK;
    __bf16* wkb = wqb + WSZ;
    __bf16* wvb = wkb + WSZ;
    __bf16* wob = wvb + WSZ;

    // 0) one-shot f32 -> bf16 conversions (removes cvt flood from GEMM loops,
    //    halves GEMM-side HBM/L2 read bytes)
    cvt_bf16<<<(int)(TOK / 8 / 256), 256, 0, stream>>>(x,  xb,  (int)(TOK / 8));
    cvt_bf16<<<(int)(WSZ / 8 / 256), 256, 0, stream>>>(Wq, wqb, (int)(WSZ / 8));
    cvt_bf16<<<(int)(WSZ / 8 / 256), 256, 0, stream>>>(Wk, wkb, (int)(WSZ / 8));
    cvt_bf16<<<(int)(WSZ / 8 / 256), 256, 0, stream>>>(Wv, wvb, (int)(WSZ / 8));
    cvt_bf16<<<(int)(WSZ / 8 / 256), 256, 0, stream>>>(Wo, wob, (int)(WSZ / 8));

    // 1) QKV projections: M=4096, N=1024, 64x128 block tiles
    qkv_proj<<<dim3(D_ / 128, (B_ * S_) / 64, 3), 128, 0, stream>>>(
        xb, wqb, wkb, wvb, qb, kb, vb);

    // 2) causal flash attention
    attn<<<dim3(S_ / 64, H_, B_), 128, 0, stream>>>(qb, kb, vb, cb);

    // 3) output projection + bias
    out_proj<<<dim3(D_ / 128, (B_ * S_) / 64), 128, 0, stream>>>(cb, wob, bo, out);
}
